// Model_55070070669362
// MI455X (gfx1250) — matrix operand order repaired and checked by reading, not compiled, not run on hardware
//
#include <hip/hip_runtime.h>

// ---------------- model constants ----------------
#define B__    32
#define L__    1024
#define CIN    3
#define DM     256          // D_MODEL
#define DI     512          // D_INNER
#define DSN    16           // D_STATE
#define DCONV  4
#define DTRANK 16
#define NH     8
#define NC     10
#define M_TOT  (B__ * L__)  // 32768 rows

typedef __attribute__((ext_vector_type(16))) _Float16 v16h;
typedef __attribute__((ext_vector_type(8)))  _Float16 v8h;
typedef __attribute__((ext_vector_type(8)))  float    v8f;
typedef __attribute__((ext_vector_type(4)))  unsigned int v4u;
typedef __attribute__((ext_vector_type(8)))  int      v8i;
typedef __attribute__((ext_vector_type(4)))  int      v4i;

// ---------------- TDM: 2D tile (rows x cols f16) global -> LDS ----------------
// D# layout per CDNA5 ISA (group0 + group1, groups 2/3 unused)
__device__ __forceinline__ void tdm_load_2d_f16(const _Float16* gsrc, unsigned lds_off,
                                                int rows, int cols, int stride_elems) {
  unsigned long long ga = (unsigned long long)(const void*)gsrc;
  v4u g0;
  g0[0] = 1u;                                      // count=1, user mode
  g0[1] = lds_off;                                 // lds_addr
  g0[2] = (unsigned)(ga & 0xFFFFFFFFull);          // global_addr[31:0]
  g0[3] = (unsigned)((ga >> 32) & 0x1FFFFFFull)    // global_addr[56:32]
        | (2u << 30);                              // type = 2 ("image")
  v8i g1;
  g1[0] = (int)(1u << 16);                         // data_size = 1 (2 bytes/elem)
  g1[1] = (int)(((unsigned)cols & 0xFFFFu) << 16); // tensor_dim0[15:0] @ [31:16]
  g1[2] = (int)((((unsigned)cols >> 16) & 0xFFFFu)
        | (((unsigned)rows & 0xFFFFu) << 16));     // dim0 hi16 | dim1 lo16
  g1[3] = (int)(((unsigned)cols & 0xFFFFu) << 16); // dim1 hi16=0 | tile_dim0 = cols
  g1[4] = (int)((unsigned)rows & 0xFFFFu);         // tile_dim1 = rows, tile_dim2 = 0
  g1[5] = (int)stride_elems;                       // tensor_dim0_stride[31:0]
  g1[6] = 0;                                       // stride hi16 | dim1_stride lo16
  g1[7] = 0;
  v4i z4 = {};
  v8i z8 = {};
  // 6-arg toolchain form: (g0, g1, g2, g3, extra, cpol)
  __builtin_amdgcn_tensor_load_to_lds(g0, g1, z4, z4, z8, 0);
}

// ---------------- WMMA fragment loaders (CDNA5 ISA 7.12.2) ----------------
// A (16x32 f16) from a row-major tile with leading dim lda: two contiguous v8h chunks.
__device__ __forceinline__ v16h load_a_frag(const _Float16* sA, int lda, int kof) {
  const int lane = threadIdx.x & 31;
  const int m    = lane & 15;
  const int khi  = (lane >> 4) << 4;               // 0 or 16
  const _Float16* base = sA + m * lda + kof + khi;
  v8h lo = *(const v8h*)(base);                    // K = khi+0..7
  v8h hi = *(const v8h*)(base + 8);                // K = khi+8..15
  v16h a;
#pragma unroll
  for (int i = 0; i < 8; ++i) { a[i] = lo[i]; a[8 + i] = hi[i]; }
  return a;
}

// B = W^T, W (N,K) row-major, caller pre-offsets by n0*ldw. Contiguous 16 halves.
__device__ __forceinline__ v16h load_b_frag(const _Float16* W, int ldw, int kof) {
  const int lane = threadIdx.x & 31;
  const int n    = lane & 15;
  const int klo  = (lane >> 4) << 4;               // 0 or 16
  return *(const v16h*)(W + (size_t)n * ldw + kof + klo);
}

// ---------------- TDM-staged WMMA GEMM: C[M,N] = A[M,K] * W[N,K]^T ----------------
// Block owns one 16-row A tile (TDM -> LDS, shared by all waves); each wave computes
// NT consecutive 16x16 N-tiles, reusing one A fragment per K-step for NT WMMAs.
// grid.x = M/16 ; grid.y * (blockDim/32) * NT = N/16.
template <int NT>
__global__ void gemm_wmma_tdm(const _Float16* __restrict__ A, int lda,
                              const _Float16* __restrict__ W, int ldw,
                              float* __restrict__ C, int ldc, int K) {
  __shared__ _Float16 sA[16 * 512];                // up to K=512
  const int nwaves = blockDim.x >> 5;
  const int wave   = threadIdx.x >> 5;
  const int m0     = blockIdx.x * 16;

  if (wave == 0) {
    tdm_load_2d_f16(A + (size_t)m0 * lda, (unsigned)(uintptr_t)(void*)sA, 16, K, lda);
    __builtin_amdgcn_s_wait_tensorcnt(0);
  }
  __syncthreads();

  const int ntile0 = (blockIdx.y * nwaves + wave) * NT;
  const _Float16* Wb[NT];
#pragma unroll
  for (int t = 0; t < NT; ++t) Wb[t] = W + (size_t)(ntile0 + t) * 16 * ldw;

  v8f acc[NT];
#pragma unroll
  for (int t = 0; t < NT; ++t) acc[t] = (v8f){};

  for (int kof = 0; kof < K; kof += 32) {
    if (kof + 32 < K) __builtin_prefetch(Wb[0] + ((threadIdx.x & 15) * ldw + kof + 32), 0, 1);
    v16h a = load_a_frag(sA, K, kof);
#pragma unroll
    for (int t = 0; t < NT; ++t) {
      v16h b = load_b_frag(Wb[t], ldw, kof);
      acc[t] = __builtin_amdgcn_wmma_f32_16x16x32_f16(false, a, false, b, (short)0,
                                                      acc[t], false, false);
    }
  }
  const int lane = threadIdx.x & 31;
  const int n  = lane & 15;
  const int mb = (lane >> 4) << 3;
#pragma unroll
  for (int t = 0; t < NT; ++t) {
    const int n0 = (ntile0 + t) * 16;
#pragma unroll
    for (int r = 0; r < 8; ++r)
      C[(size_t)(m0 + mb + r) * ldc + n0 + n] = acc[t][r];
  }
}

// ---------------- dt GEMM (K zero-padded 16->32 in the operands) + bias + softplus ----
__global__ void gemm_dt_softplus(const _Float16* __restrict__ dtr_pad,   // (M,32)
                                 const _Float16* __restrict__ w_dt_pad,  // (512,32)
                                 const float* __restrict__ dt_bias,
                                 float* __restrict__ dtv) {
  const int wave  = threadIdx.x >> 5;
  const int ntile = blockIdx.y * (blockDim.x >> 5) + wave;   // 32 n-tiles
  const int m0 = blockIdx.x * 16;
  const int n0 = ntile * 16;
  v8f acc = {};
  v16h a = load_a_frag(dtr_pad + (size_t)m0 * 32, 32, 0);
  v16h b = load_b_frag(w_dt_pad + (size_t)n0 * 32, 32, 0);
  acc = __builtin_amdgcn_wmma_f32_16x16x32_f16(false, a, false, b, (short)0, acc,
                                               false, false);
  const int lane = threadIdx.x & 31;
  const int n  = lane & 15;
  const int mb = (lane >> 4) << 3;
#pragma unroll
  for (int r = 0; r < 8; ++r) {
    float v = acc[r] + dt_bias[n0 + n];
    float sp = (v > 20.0f) ? v : __logf(1.0f + __expf(v));   // fast softplus
    dtv[(size_t)(m0 + mb + r) * DI + n0 + n] = sp;
  }
}

// ---------------- small converts ----------------
__global__ void f32_to_f16_k(const float* __restrict__ src, _Float16* __restrict__ dst,
                             int n) {
  int i = blockIdx.x * blockDim.x + threadIdx.x;
  if (i < n) dst[i] = (_Float16)src[i];
}

// src (rows, src_cols) f32 -> dst (rows, dst_cols) f16, zero-padded columns
__global__ void f32_to_f16_pad_k(const float* __restrict__ src, _Float16* __restrict__ dst,
                                 int rows, int src_cols, int dst_cols) {
  int i = blockIdx.x * blockDim.x + threadIdx.x;
  if (i < rows * dst_cols) {
    int r = i / dst_cols, c = i % dst_cols;
    dst[i] = (c < src_cols) ? (_Float16)src[(size_t)r * src_cols + c] : (_Float16)0.0f;
  }
}

// dbl (M,48) f32 cols 0..15 -> dtr_pad (M,32) f16, cols 16..31 zeroed
__global__ void dtr_pad_f16_k(const float* __restrict__ dbl, _Float16* __restrict__ dtr_pad) {
  int i = blockIdx.x * blockDim.x + threadIdx.x;   // M_TOT*32
  if (i < M_TOT * 32) {
    int m = i >> 5, c = i & 31;
    dtr_pad[i] = (c < DTRANK) ? (_Float16)dbl[(size_t)m * 48 + c] : (_Float16)0.0f;
  }
}

// ---------------- conv embedding + positional encoding ----------------
__global__ void embed_kernel(const float* __restrict__ x_enc,
                             const float* __restrict__ wconv,
                             float* __restrict__ x, _Float16* __restrict__ x_h) {
  const int row = blockIdx.x;               // b*L + l
  const int b = row >> 10, l = row & 1023;
  const int d = threadIdx.x;                // 256
  float acc = 0.0f;
#pragma unroll
  for (int k = 0; k < 3; ++k) {
    int ls = l + k - 1;
    ls = ls < 0 ? 0 : (ls > (L__ - 1) ? (L__ - 1) : ls);   // edge pad
    const float* xe = x_enc + ((size_t)b * L__ + ls) * CIN;
#pragma unroll
    for (int c = 0; c < 3; ++c) acc += xe[c] * wconv[(d * CIN + c) * 3 + k];
  }
  int i2 = d & ~1;
  float freq = __expf(-logf(10000.0f) * (float)i2 / (float)DM);
  float ang  = (float)l * freq;
  acc += (d & 1) ? __cosf(ang) : __sinf(ang);
  x[(size_t)row * DM + d]   = acc;
  x_h[(size_t)row * DM + d] = (_Float16)acc;
}

// ---------------- depthwise causal conv1d + SiLU ----------------
__global__ void dwconv_kernel(const float* __restrict__ xz,
                              const float* __restrict__ w,
                              const float* __restrict__ bias,
                              float* __restrict__ xc, _Float16* __restrict__ xc_h) {
  const int row = blockIdx.x;               // b*L + l
  const int l = row & 1023;
  const int d = threadIdx.x;                // 512
  float acc = bias[d];
#pragma unroll
  for (int k = 0; k < DCONV; ++k) {
    int lp = l - (DCONV - 1) + k;
    float xv = (lp >= 0) ? xz[((size_t)row + k - (DCONV - 1)) * (2 * DI) + d] : 0.0f;
    acc += xv * w[d * DCONV + k];
  }
  float sv = acc / (1.0f + __expf(-acc));
  xc[(size_t)row * DI + d]   = sv;
  xc_h[(size_t)row * DI + d] = (_Float16)sv;
}

// ---------------- selective scan (sequential over L) ----------------
__global__ void scan_kernel(const float* __restrict__ dtv,
                            const float* __restrict__ dbl,
                            const float* __restrict__ xc,
                            const float* __restrict__ A_log,
                            float* __restrict__ ys) {
  const int b = blockIdx.x;                 // 32
  const int d = threadIdx.x;                // 512
  float Arow[DSN], h[DSN];
#pragma unroll
  for (int n = 0; n < DSN; ++n) {
    Arow[n] = -__expf(A_log[(size_t)d * DSN + n]);
    h[n] = 0.0f;
  }
  __shared__ float sB[2][DSN], sC[2][DSN];
  const size_t base = (size_t)b * L__;
  for (int l = 0; l < L__; ++l) {
    const int pb = l & 1;
    const size_t row = base + l;
    if (threadIdx.x < 16)
      sB[pb][threadIdx.x] = dbl[row * 48 + DTRANK + threadIdx.x];
    else if (threadIdx.x < 32)
      sC[pb][threadIdx.x - 16] = dbl[row * 48 + DTRANK + DSN + (threadIdx.x - 16)];
    __syncthreads();
    const float dtc = dtv[row * DI + d];
    const float xv  = xc[row * DI + d];
    float y = 0.0f;
#pragma unroll
    for (int n = 0; n < DSN; ++n) {
      float dA = __expf(dtc * Arow[n]);
      h[n] = dA * h[n] + dtc * sB[pb][n] * xv;
      y += h[n] * sC[pb][n];
    }
    ys[row * DI + d] = y;
  }
}

// ---------------- y = (ys + xc*D) * silu(z) ----------------
__global__ void gate_kernel(const float* __restrict__ ys, const float* __restrict__ xc,
                            const float* __restrict__ xz, const float* __restrict__ Dv,
                            _Float16* __restrict__ y_h) {
  const size_t row = blockIdx.x;
  const int d = threadIdx.x;                // 512
  float v = ys[row * DI + d] + xc[row * DI + d] * Dv[d];
  float z = xz[row * (2 * DI) + DI + d];
  v *= z / (1.0f + __expf(-z));
  y_h[row * DI + d] = (_Float16)v;
}

// ---------------- LayerNorm + SiLU + classifier & attention heads ----------------
__global__ void ln_heads_kernel(const float* __restrict__ mraw,
                                const float* __restrict__ ln_g,
                                const float* __restrict__ ln_b,
                                const float* __restrict__ out_w,
                                const float* __restrict__ attn_w,
                                const float* __restrict__ attn_b,
                                const float* __restrict__ x_mark,
                                float* __restrict__ logitb,
                                float* __restrict__ smaxb) {
  __shared__ float red[DM];
  __shared__ float sm[DM];
  __shared__ float ss[NH];
  const size_t row = blockIdx.x;
  const int t = threadIdx.x;                // 256
  float v = mraw[row * DM + t];
  red[t] = v; __syncthreads();
  for (int s = DM / 2; s > 0; s >>= 1) { if (t < s) red[t] += red[t + s]; __syncthreads(); }
  const float mu = red[0] / (float)DM;
  __syncthreads();
  const float dv = v - mu;
  red[t] = dv * dv; __syncthreads();
  for (int s = DM / 2; s > 0; s >>= 1) { if (t < s) red[t] += red[t + s]; __syncthreads(); }
  const float var = red[0] / (float)DM;
  float nv = dv * rsqrtf(var + 1e-5f) * ln_g[t] + ln_b[t];
  sm[t] = nv / (1.0f + __expf(-nv));        // silu
  __syncthreads();
  if (t < NC) {
    float acc = 0.0f;
    for (int dd = 0; dd < DM; ++dd) acc += sm[dd] * out_w[t * DM + dd];
    logitb[row * NC + t] = acc * x_mark[row];
  } else if (t < NC + NH) {
    const int hh = t - NC;
    float acc = 0.0f;
    for (int dd = 0; dd < DM; ++dd) acc += sm[dd] * attn_w[hh * DM + dd];
    ss[hh] = acc + attn_b[hh];
  }
  __syncthreads();
  if (t == 0) {
    float mx = ss[0];
#pragma unroll
    for (int hh = 1; hh < NH; ++hh) mx = fmaxf(mx, ss[hh]);
    smaxb[row] = mx;
  }
}

// ---------------- softmax over L + weighted pool ----------------
__global__ void final_pool_kernel(const float* __restrict__ logitb,
                                  const float* __restrict__ smaxb,
                                  float* __restrict__ out) {
  __shared__ float red[256];
  __shared__ float pc[NC][256];
  const int b = blockIdx.x, t = threadIdx.x;
  float mx = -1e30f;
  for (int l = t; l < L__; l += 256) mx = fmaxf(mx, smaxb[b * L__ + l]);
  red[t] = mx; __syncthreads();
  for (int s = 128; s > 0; s >>= 1) { if (t < s) red[t] = fmaxf(red[t], red[t + s]); __syncthreads(); }
  const float gmax = red[0];
  __syncthreads();
  float se = 0.0f, acc[NC];
#pragma unroll
  for (int c = 0; c < NC; ++c) acc[c] = 0.0f;
  for (int l = t; l < L__; l += 256) {
    float w = __expf(smaxb[b * L__ + l] - gmax);
    se += w;
#pragma unroll
    for (int c = 0; c < NC; ++c) acc[c] += w * logitb[(size_t)(b * L__ + l) * NC + c];
  }
  red[t] = se; __syncthreads();
  for (int s = 128; s > 0; s >>= 1) { if (t < s) red[t] += red[t + s]; __syncthreads(); }
  const float denom = red[0];
#pragma unroll
  for (int c = 0; c < NC; ++c) pc[c][t] = acc[c];
  __syncthreads();
  for (int s = 128; s > 0; s >>= 1) {
    if (t < s)
#pragma unroll
      for (int c = 0; c < NC; ++c) pc[c][t] += pc[c][t + s];
    __syncthreads();
  }
  if (t < NC) out[b * NC + t] = pc[t][0] / denom;
}

// ---------------- host launcher ----------------
extern "C" void kernel_launch(void* const* d_in, const int* in_sizes, int n_in,
                              void* d_out, int out_size, void* d_ws, size_t ws_size,
                              hipStream_t stream) {
  (void)in_sizes; (void)n_in; (void)out_size; (void)ws_size;
  const float* x_enc      = (const float*)d_in[0];
  const float* x_mark     = (const float*)d_in[1];
  const float* conv_emb_w = (const float*)d_in[2];
  const float* in_proj_w  = (const float*)d_in[3];
  const float* conv1d_w   = (const float*)d_in[4];
  const float* conv1d_b   = (const float*)d_in[5];
  const float* x_proj_w   = (const float*)d_in[6];
  const float* dt_proj_w  = (const float*)d_in[7];
  const float* dt_proj_b  = (const float*)d_in[8];
  const float* A_log      = (const float*)d_in[9];
  const float* Dvec       = (const float*)d_in[10];
  const float* out_proj_w = (const float*)d_in[11];
  const float* ln_g       = (const float*)d_in[12];
  const float* ln_b       = (const float*)d_in[13];
  const float* out_w      = (const float*)d_in[14];
  const float* attn_w     = (const float*)d_in[15];
  const float* attn_b     = (const float*)d_in[16];
  float* out = (float*)d_out;

  char* p = (char*)d_ws;
  auto alloc = [&](size_t bytes) -> void* {
    void* r = (void*)p;
    p += (bytes + 255) & ~(size_t)255;
    return r;
  };
  const size_t M = M_TOT;
  float*    x       = (float*)   alloc(M * DM * 4);
  _Float16* x_h     = (_Float16*)alloc(M * DM * 2);
  _Float16* w_in_h  = (_Float16*)alloc((size_t)2 * DI * DM * 2);
  _Float16* w_xp_h  = (_Float16*)alloc((size_t)48 * DI * 2);
  _Float16* w_dt_h  = (_Float16*)alloc((size_t)DI * 32 * 2);     // padded (512,32)
  _Float16* w_out_h = (_Float16*)alloc((size_t)DM * DI * 2);
  float*    xz      = (float*)   alloc(M * 2 * DI * 4);
  float*    xc      = (float*)   alloc(M * DI * 4);
  _Float16* xc_h    = (_Float16*)alloc(M * DI * 2);
  float*    dbl     = (float*)   alloc(M * 48 * 4);
  _Float16* dtr_pad = (_Float16*)alloc(M * 32 * 2);              // padded (M,32)
  float*    dtv     = (float*)   alloc(M * DI * 4);
  float*    ysb     = (float*)   alloc(M * DI * 4);
  _Float16* y_h     = (_Float16*)alloc(M * DI * 2);
  float*    mraw    = (float*)   alloc(M * DM * 4);
  float*    logitb  = (float*)   alloc(M * NC * 4);
  float*    smaxb   = (float*)   alloc(M * 4);

  // weight converts
  f32_to_f16_k<<<(2 * DI * DM + 255) / 256, 256, 0, stream>>>(in_proj_w, w_in_h, 2 * DI * DM);
  f32_to_f16_k<<<(48 * DI + 255) / 256, 256, 0, stream>>>(x_proj_w, w_xp_h, 48 * DI);
  f32_to_f16_pad_k<<<(DI * 32 + 255) / 256, 256, 0, stream>>>(dt_proj_w, w_dt_h,
                                                              DI, DTRANK, 32);
  f32_to_f16_k<<<(DM * DI + 255) / 256, 256, 0, stream>>>(out_proj_w, w_out_h, DM * DI);

  // 1) conv embedding + pos emb
  embed_kernel<<<M_TOT, DM, 0, stream>>>(x_enc, conv_emb_w, x, x_h);

  // 2) in_proj: (32768 x 1024) = x_h(32768x256) * W(1024x256)^T   [TDM + WMMA, NT=4]
  gemm_wmma_tdm<4><<<dim3(M_TOT / 16, 2), 256, 0, stream>>>(
      x_h, DM, w_in_h, DM, xz, 2 * DI, DM);

  // 3) depthwise causal conv + silu
  dwconv_kernel<<<M_TOT, DI, 0, stream>>>(xz, conv1d_w, conv1d_b, xc, xc_h);

  // 4) x_proj: (32768 x 48) = xc_h(32768x512) * W(48x512)^T       [TDM + WMMA, NT=1]
  gemm_wmma_tdm<1><<<dim3(M_TOT / 16, 1), 96, 0, stream>>>(
      xc_h, DI, w_xp_h, DI, dbl, 48, DI);

  // 5) dt_proj + softplus                                          [WMMA]
  dtr_pad_f16_k<<<(M_TOT * 32 + 255) / 256, 256, 0, stream>>>(dbl, dtr_pad);
  gemm_dt_softplus<<<dim3(M_TOT / 16, DI / (16 * 8)), 256, 0, stream>>>(
      dtr_pad, w_dt_h, dt_proj_b, dtv);

  // 6) selective scan
  scan_kernel<<<B__, DI, 0, stream>>>(dtv, dbl, xc, A_log, ysb);

  // 7) gate
  gate_kernel<<<M_TOT, DI, 0, stream>>>(ysb, xc, xz, Dvec, y_h);

  // 8) out_proj: (32768 x 256) = y_h(32768x512) * W(256x512)^T    [TDM + WMMA, NT=2]
  gemm_wmma_tdm<2><<<dim3(M_TOT / 16, 1), 256, 0, stream>>>(
      y_h, DI, w_out_h, DI, mraw, DM, DI);

  // 9) LN + silu + heads
  ln_heads_kernel<<<M_TOT, DM, 0, stream>>>(mraw, ln_g, ln_b, out_w, attn_w, attn_b,
                                            x_mark, logitb, smaxb);

  // 10) softmax over L + pool -> (32,10)
  final_pool_kernel<<<B__, 256, 0, stream>>>(logitb, smaxb, out);
}
